// custom_attn_25048249270470
// MI455X (gfx1250) — matrix operand order repaired and checked by reading, not compiled, not run on hardware
//
#include <hip/hip_runtime.h>

// ---------------------------------------------------------------------------
// MHA forward for MI455X (gfx1250, wave32, WMMA).
// B=2, S=2048, F=1024, H=16, DH=64.
// bf16 WMMA (16x16x32, f32 accum); footprint fits in 192MB L2 so global loads
// are L2-backed. Hot loops: 2-stage ping-pong, UNIFORM body (no tail peel, no
// guards) -- next stage is prefetched unconditionally; the one-stage over-read
// on the last iteration lands in a 64KiB workspace pad and is never consumed.
// ---------------------------------------------------------------------------

constexpr int Bn  = 2;
constexpr int Sn  = 2048;
constexpr int Fn  = 1024;
constexpr int Hn  = 16;
constexpr int DHn = 64;

typedef __attribute__((ext_vector_type(16))) __bf16          v16bf;
typedef __attribute__((ext_vector_type(8)))  float           v8f;
typedef __attribute__((ext_vector_type(8)))  unsigned short  v8us;
typedef __attribute__((ext_vector_type(16))) unsigned short  v16us;

// float -> bf16 bits, round-to-nearest-even
__device__ __forceinline__ unsigned short f2bf(float f) {
  unsigned int u = __float_as_uint(f);
  unsigned int r = u + 0x7FFFu + ((u >> 16) & 1u);
  return (unsigned short)(r >> 16);
}

// ---------------------------------------------------------------------------
// WMMA fragment loaders (ISA 7.12.2, 16-bit operand layouts, wave32)
// A 16x32 (MxK): lane: m = l&15, h = l>>4; K = {8h..8h+7} U {16+8h..16+8h+7}.
// B 32x16 (KxN): lane: n = l&15, h = l>>4; K = {8h..8h+7} U {16+8h..16+8h+7}.
// ---------------------------------------------------------------------------
__device__ __forceinline__ v16bf load_frag_a(const unsigned short* __restrict__ base,
                                             int ld, int lane) {
  const int h = (lane >> 4) & 1;
  const int m = lane & 15;
  const unsigned short* p = base + (size_t)m * ld + 8 * h;
  v8us lo = *(const v8us*)(p);
  v8us hi = *(const v8us*)(p + 16);
  v16us r;
#pragma unroll
  for (int i = 0; i < 8; ++i) { r[i] = lo[i]; r[i + 8] = hi[i]; }
  return __builtin_bit_cast(v16bf, r);
}

__device__ __forceinline__ v16bf load_frag_b(const unsigned short* __restrict__ base,
                                             int ld, int lane) {
  const int h = (lane >> 4) & 1;
  const int n = lane & 15;
  const unsigned short* p = base + (size_t)n * ld + 8 * h;
  v8us lo = *(const v8us*)(p);
  v8us hi = *(const v8us*)(p + 16);
  v16us r;
#pragma unroll
  for (int i = 0; i < 8; ++i) { r[i] = lo[i]; r[i + 8] = hi[i]; }
  return __builtin_bit_cast(v16bf, r);
}

__device__ __forceinline__ v8f wmma_bf16(v16bf a, v16bf b, v8f c) {
  return __builtin_amdgcn_wmma_f32_16x16x32_bf16(false, a, false, b,
                                                 (short)0, c, false, false);
}

// ---------------------------------------------------------------------------
// Kernel 1: fp32 -> bf16 conversion (grid-stride)
// ---------------------------------------------------------------------------
__global__ void cvt_f32_bf16(const float* __restrict__ in,
                             unsigned short* __restrict__ out, int n) {
  int i = blockIdx.x * blockDim.x + threadIdx.x;
  const int stride = gridDim.x * blockDim.x;
  for (; i < n; i += stride) out[i] = f2bf(in[i]);
}

// ---------------------------------------------------------------------------
// Kernel 2: Y = X @ W^T + bias.  X:[4096,1024] bf16, W:[1024,1024] bf16.
// One wave computes a 32x64 tile (2 A-frags x 4 B-frags = 8 WMMAs per K-step).
// Uniform 2-stage ping-pong loop, 16 iterations x 64 K, no tail.
// mode 0: bf16 out [B,H,S,DH] | mode 1: bf16 out [B,H,DH,S] | mode 2: f32 flat
// ---------------------------------------------------------------------------
__global__ __launch_bounds__(256)
void gemm_bf16(const unsigned short* __restrict__ A,
               const unsigned short* __restrict__ W,
               const float* __restrict__ bias,
               unsigned short* __restrict__ outBf,
               float* __restrict__ outF,
               int mode) {
  const int lane = threadIdx.x & 31;
  const int wave = threadIdx.x >> 5;
  const int gid  = blockIdx.x * 8 + wave;       // 2048 wave-tiles
  const int NT   = Fn / 64;                     // 16 column tiles
  const int row0 = (gid / NT) * 32;
  const int col0 = (gid % NT) * 64;

  const unsigned short* A0 = A + (size_t)row0 * Fn;
  const unsigned short* A1 = A + (size_t)(row0 + 16) * Fn;

  v8f acc[2][4] = {};
  v16bf a0[2], a1[2], bb[2][4];

  auto loadStage = [&](int st, int k) {
    a0[st] = load_frag_a(A0 + k, Fn, lane);
    a1[st] = load_frag_a(A1 + k, Fn, lane);
#pragma unroll
    for (int f = 0; f < 4; ++f)
      bb[st][f] = load_frag_b(W + (size_t)(col0 + f * 16) * Fn + k, Fn, lane);
  };
  auto mmaStage = [&](int st) {
#pragma unroll
    for (int f = 0; f < 4; ++f) {
      acc[0][f] = wmma_bf16(a0[st], bb[st][f], acc[0][f]);
      acc[1][f] = wmma_bf16(a1[st], bb[st][f], acc[1][f]);
    }
  };

  loadStage(0, 0);
  for (int k0 = 0; k0 < Fn; k0 += 64) {   // uniform body, no tail
    loadStage(1, k0 + 32);
    mmaStage(0);
    loadStage(0, k0 + 64);                // last iter over-reads into pad (unused)
    mmaStage(1);
  }

  const int half = lane >> 4;
  const int n    = lane & 15;
#pragma unroll
  for (int pr = 0; pr < 2; ++pr) {
#pragma unroll
    for (int f = 0; f < 4; ++f) {
      const int c  = col0 + f * 16 + n;
      const float bv = bias[c];
#pragma unroll
      for (int r = 0; r < 8; ++r) {
        const int g = row0 + pr * 16 + r + 8 * half;   // row in [0, B*S)
        const float v = acc[pr][f][r] + bv;
        if (mode == 2) {
          outF[(size_t)g * Fn + c] = v;
        } else {
          const int bI = g / Sn, s = g % Sn;
          const int h = c / DHn, d = c % DHn;
          const size_t idx = (mode == 0)
              ? (((size_t)(bI * Hn + h)) * Sn + s) * DHn + d
              : (((size_t)(bI * Hn + h)) * DHn + d) * Sn + s;
          outBf[idx] = f2bf(v);
        }
      }
    }
  }
}

// ---------------------------------------------------------------------------
// Kernel 3: flash attention. One wave owns a 16-row q tile of one (b,h).
// Uniform 64-key ping-pong loop (two 32-key steps per iteration, no tail);
// K-fragments prefetched a full step ahead, V-fragments loaded at step start
// so softmax VALU hides them, probs transposed C->A via per-wave LDS.
// ---------------------------------------------------------------------------
__global__ __launch_bounds__(256)
void attn_fwd(const unsigned short* __restrict__ qh,  // [B,H,S,DH] bf16
              const unsigned short* __restrict__ kh,  // [B,H,S,DH] bf16
              const unsigned short* __restrict__ vT,  // [B,H,DH,S] bf16
              const float* __restrict__ mask,         // [S,S]
              unsigned short* __restrict__ ctx) {     // [B,S,F] bf16
  __shared__ __align__(16) unsigned short pT[8][16 * 32];

  const int lane = threadIdx.x & 31;
  const int wave = threadIdx.x >> 5;
  const int gid  = blockIdx.x * 8 + wave;             // B*H*(S/16) = 4096 tiles
  const int qt   = gid % (Sn / 16);
  const int bh   = gid / (Sn / 16);
  const int h    = bh % Hn;
  const int b    = bh / Hn;
  const int q0   = qt * 16;

  const unsigned short* qBase = qh + ((size_t)bh * Sn + q0) * DHn;
  const unsigned short* kBase = kh + (size_t)bh * Sn * DHn;
  const unsigned short* vBase = vT + (size_t)bh * DHn * Sn;
  unsigned short* myP = pT[wave];

  const v16bf qa0 = load_frag_a(qBase + 0,  DHn, lane);
  const v16bf qa1 = load_frag_a(qBase + 32, DHn, lane);

  const int half = lane >> 4;
  const int n    = lane & 15;
  const float scale = 0.125f;                         // 1/sqrt(DH)

  float mrow[8], lrow[8];
  v8f acc[4] = {};
#pragma unroll
  for (int r = 0; r < 8; ++r) { mrow[r] = -1e30f; lrow[r] = 0.0f; }

  // kb[st][0..3]: key tiles j and j+16, d-chunks 0..31 / 32..63
  v16bf kb[2][4];
  auto loadK = [&](int st, int j) {
    const unsigned short* k0p = kBase + (size_t)j * DHn;
    const unsigned short* k1p = kBase + (size_t)(j + 16) * DHn;
    kb[st][0] = load_frag_b(k0p + 0,  DHn, lane);
    kb[st][1] = load_frag_b(k0p + 32, DHn, lane);
    kb[st][2] = load_frag_b(k1p + 0,  DHn, lane);
    kb[st][3] = load_frag_b(k1p + 32, DHn, lane);
  };

  // one 32-key flash step using K fragments of stage `st`
  auto step = [&](int j, int st) {
    v16bf vb[4];
#pragma unroll
    for (int f = 0; f < 4; ++f)
      vb[f] = load_frag_b(vBase + (size_t)(f * 16) * Sn + j, Sn, lane);

    v8f s0 = {}; v8f s1 = {};
    s0 = wmma_bf16(qa0, kb[st][0], s0);
    s0 = wmma_bf16(qa1, kb[st][1], s0);
    s1 = wmma_bf16(qa0, kb[st][2], s1);
    s1 = wmma_bf16(qa1, kb[st][3], s1);

#pragma unroll
    for (int r = 0; r < 8; ++r) {
      const int qrow = q0 + r + 8 * half;
      float x0 = s0[r] * scale + mask[(size_t)qrow * Sn + (j + n)];
      float x1 = s1[r] * scale + mask[(size_t)qrow * Sn + (j + 16 + n)];

      float bm = fmaxf(x0, x1);
#pragma unroll
      for (int off = 1; off < 16; off <<= 1) bm = fmaxf(bm, __shfl_xor(bm, off, 32));

      const float mNew  = fmaxf(mrow[r], bm);
      const float alpha = __expf(mrow[r] - mNew);
      mrow[r] = mNew;

      const float p0 = __expf(x0 - mNew);
      const float p1 = __expf(x1 - mNew);
      float ps = p0 + p1;
#pragma unroll
      for (int off = 1; off < 16; off <<= 1) ps += __shfl_xor(ps, off, 32);
      lrow[r] = lrow[r] * alpha + ps;

#pragma unroll
      for (int f = 0; f < 4; ++f) acc[f][r] *= alpha;

      const int mRow = r + 8 * half;
      myP[mRow * 32 + n]      = f2bf(p0);
      myP[mRow * 32 + 16 + n] = f2bf(p1);
    }

    const v16bf pa = load_frag_a(myP, 32, lane);      // probs 16x32 A-fragment
#pragma unroll
    for (int f = 0; f < 4; ++f)
      acc[f] = wmma_bf16(pa, vb[f], acc[f]);
  };

  // uniform 2-stage ping-pong, 32 iterations x 64 keys, no tail
  loadK(0, 0);
  for (int j = 0; j < Sn; j += 64) {
    loadK(1, j + 32);
    step(j, 0);
    loadK(0, j + 64);          // last iter over-reads next region (unused)
    step(j + 32, 1);
  }

  // epilogue: normalize, write context bf16 [B,S,F]
#pragma unroll
  for (int r = 0; r < 8; ++r) {
    const float inv  = 1.0f / lrow[r];
    const int   srow = q0 + r + 8 * half;
#pragma unroll
    for (int f = 0; f < 4; ++f) {
      const int c = h * DHn + f * 16 + n;
      ctx[((size_t)b * Sn + srow) * Fn + c] = f2bf(acc[f][r] * inv);
    }
  }
}

// ---------------------------------------------------------------------------
// Launch
// ---------------------------------------------------------------------------
extern "C" void kernel_launch(void* const* d_in, const int* in_sizes, int n_in,
                              void* d_out, int out_size, void* d_ws, size_t ws_size,
                              hipStream_t stream) {
  const float* Q    = (const float*)d_in[0];
  const float* K    = (const float*)d_in[1];
  const float* V    = (const float*)d_in[2];
  const float* mask = (const float*)d_in[3];
  const float* Wq   = (const float*)d_in[4];
  const float* bq   = (const float*)d_in[5];
  const float* Wk   = (const float*)d_in[6];
  const float* bk   = (const float*)d_in[7];
  const float* Wv   = (const float*)d_in[8];
  const float* bv   = (const float*)d_in[9];
  const float* Wo   = (const float*)d_in[10];
  const float* bo   = (const float*)d_in[11];

  const size_t EL  = (size_t)Bn * Sn * Fn;   // 4,194,304
  const size_t WEL = (size_t)Fn * Fn;        // 1,048,576

  unsigned short* ws  = (unsigned short*)d_ws;     // 64 MiB + 64 KiB pad
  unsigned short* Qb  = ws;
  unsigned short* Kb  = Qb  + EL;
  unsigned short* Vb  = Kb  + EL;
  unsigned short* Wqb = Vb  + EL;
  unsigned short* Wkb = Wqb + WEL;
  unsigned short* Wvb = Wkb + WEL;
  unsigned short* Wob = Wvb + WEL;
  unsigned short* qhB = Wob + WEL;   // [B,H,S,DH]
  unsigned short* khB = qhB + EL;    // [B,H,S,DH]
  unsigned short* vTB = khB + EL;    // [B,H,DH,S]
  unsigned short* ctx = vTB + EL;    // [B,S,F]; followed by 64KiB pad for
                                     // the unconditional last-stage prefetch

  // 1) fp32 -> bf16
  cvt_f32_bf16<<<1024, 256, 0, stream>>>(Q,  Qb,  (int)EL);
  cvt_f32_bf16<<<1024, 256, 0, stream>>>(K,  Kb,  (int)EL);
  cvt_f32_bf16<<<1024, 256, 0, stream>>>(V,  Vb,  (int)EL);
  cvt_f32_bf16<<<256,  256, 0, stream>>>(Wq, Wqb, (int)WEL);
  cvt_f32_bf16<<<256,  256, 0, stream>>>(Wk, Wkb, (int)WEL);
  cvt_f32_bf16<<<256,  256, 0, stream>>>(Wv, Wvb, (int)WEL);
  cvt_f32_bf16<<<256,  256, 0, stream>>>(Wo, Wob, (int)WEL);

  // 2) projections (256 blocks x 8 waves = 2048 tiles of 32x64)
  gemm_bf16<<<256, 256, 0, stream>>>(Qb, Wqb, bq, qhB, nullptr, 0);
  gemm_bf16<<<256, 256, 0, stream>>>(Kb, Wkb, bk, khB, nullptr, 0);
  gemm_bf16<<<256, 256, 0, stream>>>(Vb, Wvb, bv, vTB, nullptr, 1);

  // 3) flash attention (512 blocks x 8 waves = 4096 q-tiles)
  attn_fwd<<<512, 256, 0, stream>>>(qhB, khB, vTB, mask, ctx);

  // 4) output projection -> fp32 d_out [B,S,F]
  gemm_bf16<<<256, 256, 0, stream>>>(ctx, Wob, bo, nullptr, (float*)d_out, 2);
}
